// SAGE_sup_1168231104598
// MI455X (gfx1250) — matrix operand order repaired and checked by reading, not compiled, not run on hardware
//
#include <hip/hip_runtime.h>
#include <stdint.h>

#define IN_C 128
#define HID_C 256
#define CLS_C 64

typedef __attribute__((ext_vector_type(16))) __bf16 v16bf;
typedef __attribute__((ext_vector_type(8)))  float  v8f;

// ---------- bf16 helpers (storage = unsigned short at all ABI boundaries) ----------
__device__ __forceinline__ unsigned short f2bf(float f) {
  // round-to-nearest-even; inputs are finite (gaussian data / relu / sigmoid pre-act)
  union { float f; unsigned u; } x; x.f = f;
  unsigned u = x.u + (0x7fffu + ((x.u >> 16) & 1u));
  return (unsigned short)(u >> 16);
}
__device__ __forceinline__ float bf2f(unsigned short h) {
  union { unsigned u; float f; } x; x.u = ((unsigned)h) << 16;
  return x.f;
}
__device__ __forceinline__ unsigned pack2(unsigned short lo, unsigned short hi) {
  return ((unsigned)hi << 16) | (unsigned)lo;
}

// ---------- WMMA fragment loaders (bf16 16x16x32, wave32 ISA layouts) ----------
// A 16x32 (MxK): lane L(<16) = row L holds K {base..base+7, base+16..base+23};
// lane L+16 = row L holds K {+8..+15, +24..+31}. Caller pre-offsets by (lane>>4)*8.
__device__ __forceinline__ v16bf load_a_frag(const unsigned short* __restrict__ p) {
  v16bf a;
  uint4 lo = *(const uint4*)(p);        // 8 bf16
  uint4 hi = *(const uint4*)(p + 16);   // 8 bf16 at K+16
  __builtin_memcpy(&a, &lo, 16);
  __builtin_memcpy(((char*)&a) + 16, &hi, 16);
  return a;
}
// B 32x16 (KxN): lane L(<16) = col L holds K {base..base+7, base+16..base+23};
// lane L+16 holds K {+8..+15, +24..+31}. Weights pre-transposed (N-major, K contiguous).
__device__ __forceinline__ v16bf load_b_frag(const unsigned short* __restrict__ p) {
  v16bf b;
  uint4 lo = *(const uint4*)(p);
  uint4 hi = *(const uint4*)(p + 16);
  __builtin_memcpy(&b, &lo, 16);
  __builtin_memcpy(((char*)&b) + 16, &hi, 16);
  return b;
}
#define WMMA_BF16(A, B, C) \
  __builtin_amdgcn_wmma_f32_16x16x32_bf16(false, (A), false, (B), (short)0, (C), false, false)

// ---------- utility kernels ----------
__global__ void zero_f32(float* __restrict__ p, long n) {
  long i = (long)blockIdx.x * blockDim.x + threadIdx.x;
  long stride = (long)gridDim.x * blockDim.x;
  for (; i < n; i += stride) p[i] = 0.0f;
}

// W (K x N, f32, row-major) -> Wt (N x K, bf16, row-major)
__global__ void transpose_w(const float* __restrict__ W, unsigned short* __restrict__ Wt,
                            int K, int N) {
  int idx = blockIdx.x * blockDim.x + threadIdx.x;
  if (idx >= K * N) return;
  int n = idx / K, k = idx - n * K;
  Wt[idx] = f2bf(W[(size_t)k * N + n]);
}

__global__ void degree_k(const long long* __restrict__ ei, float* __restrict__ deg, int E) {
  int e = blockIdx.x * blockDim.x + threadIdx.x;
  if (e < E) atomicAdd(&deg[(int)ei[(size_t)E + e]], 1.0f);
}

// one wave per edge; lane moves float4 of the 128-wide row; atomics resolve in L2
__global__ __launch_bounds__(256) void scatter1(const float4* __restrict__ x4,
                                                const long long* __restrict__ ei,
                                                float* __restrict__ agg1, int E) {
  int wid  = (int)((blockIdx.x * 256u + threadIdx.x) >> 5);
  int lane = (int)(threadIdx.x & 31u);
  if (wid >= E) return;
  int s = (int)ei[wid];
  int d = (int)ei[(size_t)E + wid];
  float4 v = x4[(size_t)s * (IN_C / 4) + lane];
  float* p = agg1 + (size_t)d * IN_C + lane * 4;
  atomicAdd(p + 0, v.x); atomicAdd(p + 1, v.y);
  atomicAdd(p + 2, v.z); atomicAdd(p + 3, v.w);
}

// one wave per edge; lane moves 8 bf16 of the 256-wide hidden row
__global__ __launch_bounds__(256) void scatter2(const unsigned short* __restrict__ hb,
                                                const long long* __restrict__ ei,
                                                float* __restrict__ agg2, int E) {
  int wid  = (int)((blockIdx.x * 256u + threadIdx.x) >> 5);
  int lane = (int)(threadIdx.x & 31u);
  if (wid >= E) return;
  int s = (int)ei[wid];
  int d = (int)ei[(size_t)E + wid];
  uint4 raw = *(const uint4*)(hb + (size_t)s * HID_C + lane * 8);
  unsigned short h[8];
  __builtin_memcpy(h, &raw, 16);
  float* p = agg2 + (size_t)d * HID_C + lane * 8;
#pragma unroll
  for (int i = 0; i < 8; ++i) atomicAdd(p + i, bf2f(h[i]));
}

// xb = bf16(x); agg1b = bf16(agg1 / max(deg,1))   — packed 2 elems/thread
__global__ void convert1(const float2* __restrict__ x2, const float2* __restrict__ a2,
                         const float* __restrict__ deg,
                         unsigned* __restrict__ xb2, unsigned* __restrict__ agg1b2,
                         long nPairs) {                       // nPairs = nodes*64
  long i = (long)blockIdx.x * blockDim.x + threadIdx.x;
  if (i >= nPairs) return;
  float r = 1.0f / fmaxf(deg[i >> 6], 1.0f);                  // 64 pairs per 128-wide row
  float2 xv = x2[i];
  float2 av = a2[i];
  xb2[i]    = pack2(f2bf(xv.x), f2bf(xv.y));
  agg1b2[i] = pack2(f2bf(av.x * r), f2bf(av.y * r));
}

// agg2b = bf16(agg2 / max(deg,1))  — packed 2 elems/thread
__global__ void convert2(const float2* __restrict__ a2, const float* __restrict__ deg,
                         unsigned* __restrict__ agg2b2, long nPairs) { // nodes*128
  long i = (long)blockIdx.x * blockDim.x + threadIdx.x;
  if (i >= nPairs) return;
  float r = 1.0f / fmaxf(deg[i >> 7], 1.0f);                  // 128 pairs per 256-wide row
  float2 av = a2[i];
  agg2b2[i] = pack2(f2bf(av.x * r), f2bf(av.y * r));
}

// ---------- layer 1: h = relu(agg@W1l + x@W1r + b1) ----------
// One wave per (16-row, 64-col) output block: 4 accumulators share each A fragment.
// B fragments are preloaded into distinct registers so the backend can issue one
// load clause and cover WMMA issue with partial s_wait_loadcnt.
__global__ __launch_bounds__(256) void gemm_layer1(
    const unsigned short* __restrict__ aggb, const unsigned short* __restrict__ xb,
    const unsigned short* __restrict__ Wlt,  const unsigned short* __restrict__ Wrt,
    const float* __restrict__ bias, unsigned short* __restrict__ hb, int mTiles) {
  const int lane = (int)(threadIdx.x & 31u);
  const int wave = (int)(threadIdx.x >> 5u);
  const int tile = (int)blockIdx.x * 8 + wave;
  const int nGroups = HID_C / 64;               // 4 groups of 4 N-tiles
  if (tile >= mTiles * nGroups) return;         // wave-uniform; EXEC stays all-ones
  const int mt = tile / nGroups, ng = tile - mt * nGroups;
  const int m0 = mt * 16, n0 = ng * 64;

  const int rowA = m0 + (lane & 15);
  const int aOff = (lane >> 4) * 8;
  const int colB = n0 + (lane & 15);
  const int bOff = (lane >> 4) * 8;

  v8f acc0 = {}, acc1 = {}, acc2 = {}, acc3 = {};
#pragma unroll
  for (int k0 = 0; k0 < IN_C; k0 += 32) {       // agg @ W1l
    const unsigned short* wb = Wlt + (size_t)colB * IN_C + k0 + bOff;
    v16bf a  = load_a_frag(aggb + (size_t)rowA * IN_C + k0 + aOff);
    v16bf b0 = load_b_frag(wb + (size_t)(0 * 16) * IN_C);
    v16bf b1 = load_b_frag(wb + (size_t)(1 * 16) * IN_C);
    v16bf b2 = load_b_frag(wb + (size_t)(2 * 16) * IN_C);
    v16bf b3 = load_b_frag(wb + (size_t)(3 * 16) * IN_C);
    acc0 = WMMA_BF16(a, b0, acc0);
    acc1 = WMMA_BF16(a, b1, acc1);
    acc2 = WMMA_BF16(a, b2, acc2);
    acc3 = WMMA_BF16(a, b3, acc3);
  }
#pragma unroll
  for (int k0 = 0; k0 < IN_C; k0 += 32) {       // x @ W1r
    const unsigned short* wb = Wrt + (size_t)colB * IN_C + k0 + bOff;
    v16bf a  = load_a_frag(xb + (size_t)rowA * IN_C + k0 + aOff);
    v16bf b0 = load_b_frag(wb + (size_t)(0 * 16) * IN_C);
    v16bf b1 = load_b_frag(wb + (size_t)(1 * 16) * IN_C);
    v16bf b2 = load_b_frag(wb + (size_t)(2 * 16) * IN_C);
    v16bf b3 = load_b_frag(wb + (size_t)(3 * 16) * IN_C);
    acc0 = WMMA_BF16(a, b0, acc0);
    acc1 = WMMA_BF16(a, b1, acc1);
    acc2 = WMMA_BF16(a, b2, acc2);
    acc3 = WMMA_BF16(a, b3, acc3);
  }

  const int rowD = m0 + (lane >> 4) * 8;
  v8f accs[4] = {acc0, acc1, acc2, acc3};
#pragma unroll
  for (int j = 0; j < 4; ++j) {
    const int colD = n0 + j * 16 + (lane & 15);
    const float bv = bias[colD];
#pragma unroll
    for (int i = 0; i < 8; ++i) {
      float v = accs[j][i] + bv;
      v = v > 0.0f ? v : 0.0f;                  // ReLU
      hb[(size_t)(rowD + i) * HID_C + colD] = f2bf(v);
    }
  }
}

// ---------- layer 2: out = sigmoid(agg@W2l + h@W2r + b2) ----------
// One wave produces the full 16x64 output row block (all 4 N-tiles).
__global__ __launch_bounds__(256) void gemm_layer2(
    const unsigned short* __restrict__ aggb, const unsigned short* __restrict__ hb,
    const unsigned short* __restrict__ Wlt,  const unsigned short* __restrict__ Wrt,
    const float* __restrict__ bias, float* __restrict__ out, int mTiles) {
  const int lane = (int)(threadIdx.x & 31u);
  const int wave = (int)(threadIdx.x >> 5u);
  const int mt = (int)blockIdx.x * 8 + wave;
  if (mt >= mTiles) return;                     // wave-uniform
  const int m0 = mt * 16;

  const int rowA = m0 + (lane & 15);
  const int aOff = (lane >> 4) * 8;
  const int colB = (lane & 15);
  const int bOff = (lane >> 4) * 8;

  v8f acc0 = {}, acc1 = {}, acc2 = {}, acc3 = {};
#pragma unroll
  for (int k0 = 0; k0 < HID_C; k0 += 32) {      // agg2 @ W2l
    const unsigned short* wb = Wlt + (size_t)colB * HID_C + k0 + bOff;
    v16bf a  = load_a_frag(aggb + (size_t)rowA * HID_C + k0 + aOff);
    v16bf b0 = load_b_frag(wb + (size_t)(0 * 16) * HID_C);
    v16bf b1 = load_b_frag(wb + (size_t)(1 * 16) * HID_C);
    v16bf b2 = load_b_frag(wb + (size_t)(2 * 16) * HID_C);
    v16bf b3 = load_b_frag(wb + (size_t)(3 * 16) * HID_C);
    acc0 = WMMA_BF16(a, b0, acc0);
    acc1 = WMMA_BF16(a, b1, acc1);
    acc2 = WMMA_BF16(a, b2, acc2);
    acc3 = WMMA_BF16(a, b3, acc3);
  }
#pragma unroll
  for (int k0 = 0; k0 < HID_C; k0 += 32) {      // h @ W2r
    const unsigned short* wb = Wrt + (size_t)colB * HID_C + k0 + bOff;
    v16bf a  = load_a_frag(hb + (size_t)rowA * HID_C + k0 + aOff);
    v16bf b0 = load_b_frag(wb + (size_t)(0 * 16) * HID_C);
    v16bf b1 = load_b_frag(wb + (size_t)(1 * 16) * HID_C);
    v16bf b2 = load_b_frag(wb + (size_t)(2 * 16) * HID_C);
    v16bf b3 = load_b_frag(wb + (size_t)(3 * 16) * HID_C);
    acc0 = WMMA_BF16(a, b0, acc0);
    acc1 = WMMA_BF16(a, b1, acc1);
    acc2 = WMMA_BF16(a, b2, acc2);
    acc3 = WMMA_BF16(a, b3, acc3);
  }

  const int rowD = m0 + (lane >> 4) * 8;
  v8f accs[4] = {acc0, acc1, acc2, acc3};
#pragma unroll
  for (int j = 0; j < 4; ++j) {
    const int colD = j * 16 + (lane & 15);
    const float bv = bias[colD];
#pragma unroll
    for (int i = 0; i < 8; ++i) {
      float v = accs[j][i] + bv;
      out[(size_t)(rowD + i) * CLS_C + colD] = 1.0f / (1.0f + __expf(-v)); // sigmoid
    }
  }
}

// ---------- host launcher ----------
extern "C" void kernel_launch(void* const* d_in, const int* in_sizes, int n_in,
                              void* d_out, int out_size, void* d_ws, size_t ws_size,
                              hipStream_t stream) {
  const float*     x   = (const float*)d_in[0];
  const long long* ei  = (const long long*)d_in[1];   // int64 edge_index [2, E]
  const float*     W1l = (const float*)d_in[2];
  const float*     W1r = (const float*)d_in[3];
  const float*     b1  = (const float*)d_in[4];
  const float*     W2l = (const float*)d_in[5];
  const float*     W2r = (const float*)d_in[6];
  const float*     b2  = (const float*)d_in[7];
  float*           out = (float*)d_out;

  const int nNodes = in_sizes[0] / IN_C;   // 100000
  const int E      = in_sizes[1] / 2;      // 1600000
  const int mTiles = nNodes / 16;          // 6250 (exact)

  // workspace partition (256B aligned slices)
  char* ws = (char*)d_ws;
  auto take = [&](size_t bytes) -> char* {
    char* p = ws;
    ws += (bytes + 255) & ~(size_t)255;
    return p;
  };
  float*          deg   = (float*)take((size_t)nNodes * 4);
  float*          agg1  = (float*)take((size_t)nNodes * IN_C * 4);
  float*          agg2  = (float*)take((size_t)nNodes * HID_C * 4);
  unsigned short* xb    = (unsigned short*)take((size_t)nNodes * IN_C * 2);
  unsigned short* agg1b = (unsigned short*)take((size_t)nNodes * IN_C * 2);
  unsigned short* hb    = (unsigned short*)take((size_t)nNodes * HID_C * 2);
  unsigned short* agg2b = (unsigned short*)take((size_t)nNodes * HID_C * 2);
  unsigned short* W1lt  = (unsigned short*)take((size_t)IN_C * HID_C * 2);
  unsigned short* W1rt  = (unsigned short*)take((size_t)IN_C * HID_C * 2);
  unsigned short* W2lt  = (unsigned short*)take((size_t)HID_C * CLS_C * 2);
  unsigned short* W2rt  = (unsigned short*)take((size_t)HID_C * CLS_C * 2);

  // zero accumulators (every call: deterministic)
  zero_f32<<<2048, 256, 0, stream>>>(deg,  (long)nNodes);
  zero_f32<<<2048, 256, 0, stream>>>(agg1, (long)nNodes * IN_C);
  zero_f32<<<2048, 256, 0, stream>>>(agg2, (long)nNodes * HID_C);

  // weight transpose + bf16 convert (tiny)
  transpose_w<<<(IN_C * HID_C + 255) / 256, 256, 0, stream>>>(W1l, W1lt, IN_C, HID_C);
  transpose_w<<<(IN_C * HID_C + 255) / 256, 256, 0, stream>>>(W1r, W1rt, IN_C, HID_C);
  transpose_w<<<(HID_C * CLS_C + 255) / 256, 256, 0, stream>>>(W2l, W2lt, HID_C, CLS_C);
  transpose_w<<<(HID_C * CLS_C + 255) / 256, 256, 0, stream>>>(W2r, W2rt, HID_C, CLS_C);

  // degrees
  degree_k<<<(E + 255) / 256, 256, 0, stream>>>(ei, deg, E);

  // layer 1: aggregate, normalize+convert, fused dual-GEMM + ReLU
  scatter1<<<(E + 7) / 8, 256, 0, stream>>>((const float4*)x, ei, agg1, E);
  {
    long nPairs = (long)nNodes * (IN_C / 2);
    convert1<<<(unsigned)((nPairs + 255) / 256), 256, 0, stream>>>(
        (const float2*)x, (const float2*)agg1, deg, (unsigned*)xb, (unsigned*)agg1b, nPairs);
  }
  gemm_layer1<<<(mTiles * (HID_C / 64) + 7) / 8, 256, 0, stream>>>(agg1b, xb, W1lt, W1rt, b1, hb, mTiles);

  // layer 2: aggregate h, normalize+convert, fused dual-GEMM + sigmoid
  scatter2<<<(E + 7) / 8, 256, 0, stream>>>(hb, ei, agg2, E);
  {
    long nPairs = (long)nNodes * (HID_C / 2);
    convert2<<<(unsigned)((nPairs + 255) / 256), 256, 0, stream>>>(
        (const float2*)agg2, deg, (unsigned*)agg2b, nPairs);
  }
  gemm_layer2<<<(mTiles + 7) / 8, 256, 0, stream>>>(agg2b, hb, W2lt, W2rt, b2, out, mTiles);
}
